// Henrion_MPNNConv_67388036874510
// MI455X (gfx1250) — matrix operand order repaired and checked by reading, not compiled, not run on hardware
//
#include <hip/hip_runtime.h>
#include <hip/hip_bf16.h>

// ---------------------------------------------------------------------------
// MPNN (Henrion) on gfx1250: all GEMMs via v_wmma_f32_16x16x32_bf16 (wave32),
// packed weights staged in LDS per workgroup (ds_load_b128 B-fragments),
// scatter-add via global_atomic_add_f32 (L2-resident), GRU gates as VALU.
// ---------------------------------------------------------------------------

typedef __attribute__((ext_vector_type(16))) __bf16 v16bf;
typedef __attribute__((ext_vector_type(8)))  __bf16 v8bf;
typedef __attribute__((ext_vector_type(8)))  float  v8f;

#define IN_DIM  128
#define OUT_DIM 64
#define GRU_IN  192   // OUT_DIM + IN_DIM
#define GRU_3H  192   // 3 * OUT_DIM

// ---------------------------------------------------------------------------
// Weight repack: bf16 B-fragment tiling. Tile = 32(K) x 16(N).
// B-matrix, 16-bit, wave32 (matches A loaders): half = lane>>4, elems 0..7 =
// K kt*32+8*half+0..7, elems 8..15 = K kt*32+16+8*half+0..7;
// column n = lane & 15. Packed layout: dst[((tile*32 + lane)*16) + i].
// transpose=1 builds B[k][n] = src[n*src_ld + k] (for w_ih.T / w_hh.T).
// ---------------------------------------------------------------------------
__global__ void pack_weights_kernel(const float* __restrict__ src,
                                    __bf16* __restrict__ dst,
                                    int K, int N, int src_ld, int transpose) {
    int total = (K / 32) * (N / 16) * 512;
    int p = blockIdx.x * blockDim.x + threadIdx.x;
    if (p >= total) return;
    int ntiles = N / 16;
    int tile = p >> 9;
    int l    = (p >> 4) & 31;
    int i    = p & 15;
    int kt = tile / ntiles;
    int nt = tile % ntiles;
    int k = kt * 32 + ((l >> 4) << 3) + (i & 7) + ((i >> 3) << 4);
    int n = nt * 16 + (l & 15);
    float v = transpose ? src[(size_t)n * src_ld + k]
                        : src[(size_t)k * src_ld + n];
    dst[p] = (__bf16)v;
}

__global__ void cvt_f32_to_bf16_kernel(const float* __restrict__ src,
                                       __bf16* __restrict__ dst, int n) {
    int t = blockIdx.x * blockDim.x + threadIdx.x;
    if (t < n) dst[t] = (__bf16)src[t];
}

__global__ void zero_f32_kernel(float* __restrict__ p, int n) {
    int t = blockIdx.x * blockDim.x + threadIdx.x;
    if (t < n) p[t] = 0.0f;
}

// ---------------------------------------------------------------------------
// Stage packed weights into LDS (cooperative, 16B vectors), then barrier.
// Amortizes weight reads 8x across the block's waves and turns B-fragment
// reads into short-latency ds_load_b128.
// ---------------------------------------------------------------------------
__device__ __forceinline__ void stage_weights(const __bf16* __restrict__ g,
                                              __bf16* s, int halfs) {
    int nvec = halfs >> 3;  // 8 halfs = 16 bytes per vector
    for (int i = threadIdx.x; i < nvec; i += blockDim.x) {
        ((v8bf*)s)[i] = ((const v8bf*)g)[i];
    }
    __syncthreads();
}

// ---------------------------------------------------------------------------
// Fragment loaders (wave32; lane = threadIdx.x & 31).
// A-matrix 16x32 bf16 layout (ISA 7.12.2): lane l -> m = l&15,
//   half = l>>4: elems 0..7 = K k0+8*half+0..7, elems 8..15 = +16.
// ---------------------------------------------------------------------------
__device__ __forceinline__ v16bf a_frag_bf16(const __bf16* __restrict__ base,
                                             int row0, int ld, int k0) {
    int l = threadIdx.x & 31;
    int m = l & 15;
    int koff = (l >> 4) * 8;
    const __bf16* p = base + (size_t)(row0 + m) * ld + k0 + koff;
    v8bf lo = *(const v8bf*)(p);
    v8bf hi = *(const v8bf*)(p + 16);
    return __builtin_shufflevector(lo, hi, 0, 1, 2, 3, 4, 5, 6, 7,
                                           8, 9, 10, 11, 12, 13, 14, 15);
}

// A fragment sourced from f32 (optionally relu) with on-the-fly bf16 convert.
__device__ __forceinline__ v16bf a_frag_f32(const float* __restrict__ base,
                                            int row0, int ld, int k0,
                                            bool relu) {
    int l = threadIdx.x & 31;
    int m = l & 15;
    int koff = (l >> 4) * 8;
    const float* p = base + (size_t)(row0 + m) * ld + k0 + koff;
    v16bf a;
#pragma unroll
    for (int i = 0; i < 8; ++i) {
        float v0 = p[i];
        float v1 = p[16 + i];
        if (relu) { v0 = fmaxf(v0, 0.0f); v1 = fmaxf(v1, 0.0f); }
        a[i]     = (__bf16)v0;
        a[8 + i] = (__bf16)v1;
    }
    return a;
}

// B fragment out of the LDS-staged packed weights -> ds_load_b128 x2.
__device__ __forceinline__ v16bf b_frag_lds(const __bf16* lds_base, int tile) {
    int l = threadIdx.x & 31;
    const v8bf* q = (const v8bf*)(lds_base + ((size_t)tile * 32 + l) * 16);
    v8bf b0 = q[0];
    v8bf b1 = q[1];
    return __builtin_shufflevector(b0, b1, 0, 1, 2, 3, 4, 5, 6, 7,
                                           8, 9, 10, 11, 12, 13, 14, 15);
}

// ---------------------------------------------------------------------------
// GEMM 1: out = x @ mlp_w + mlp_b   [N,128]x[128,64] -> h(f32) + out(bf16)
// One wave per 16-row block; 4 K-chunks, 4 N-tiles. Weights in LDS (16KB).
// ---------------------------------------------------------------------------
__global__ __launch_bounds__(256)
void mlp_gemm_kernel(const __bf16* __restrict__ xb,
                     const __bf16* __restrict__ wpk,
                     const float* __restrict__ bias,
                     float* __restrict__ h,
                     __bf16* __restrict__ outb, int rowblocks) {
    extern __shared__ __bf16 smem[];
    stage_weights(wpk, smem, IN_DIM * OUT_DIM);
    int wave = (blockIdx.x * blockDim.x + threadIdx.x) >> 5;
    if (wave >= rowblocks) return;
    int row0 = wave * 16;
    v8f acc[4] = {};
#pragma unroll
    for (int kt = 0; kt < 4; ++kt) {
        v16bf a = a_frag_bf16(xb, row0, IN_DIM, kt * 32);
#pragma unroll
        for (int nt = 0; nt < 4; ++nt) {
            v16bf b = b_frag_lds(smem, kt * 4 + nt);
            acc[nt] = __builtin_amdgcn_wmma_f32_16x16x32_bf16(
                false, a, false, b, (short)0, acc[nt], false, false);
        }
    }
    int l = threadIdx.x & 31;
    int n = l & 15;
    int mo = (l >> 4) * 8;
#pragma unroll
    for (int nt = 0; nt < 4; ++nt) {
        float bv = bias[nt * 16 + n];
#pragma unroll
        for (int r = 0; r < 8; ++r) {
            float v = acc[nt][r] + bv;
            size_t idx = (size_t)(row0 + mo + r) * OUT_DIM + nt * 16 + n;
            h[idx] = v;
            outb[idx] = (__bf16)v;
        }
    }
}

// ---------------------------------------------------------------------------
// GEMM 2: msg = out @ conv_w + conv_b   [N,64]x[64,64] -> f32. LDS 8KB.
// ---------------------------------------------------------------------------
__global__ __launch_bounds__(256)
void conv_gemm_kernel(const __bf16* __restrict__ outb,
                      const __bf16* __restrict__ wpk,
                      const float* __restrict__ bias,
                      float* __restrict__ msg, int rowblocks) {
    extern __shared__ __bf16 smem[];
    stage_weights(wpk, smem, OUT_DIM * OUT_DIM);
    int wave = (blockIdx.x * blockDim.x + threadIdx.x) >> 5;
    if (wave >= rowblocks) return;
    int row0 = wave * 16;
    v8f acc[4] = {};
#pragma unroll
    for (int kt = 0; kt < 2; ++kt) {
        v16bf a = a_frag_bf16(outb, row0, OUT_DIM, kt * 32);
#pragma unroll
        for (int nt = 0; nt < 4; ++nt) {
            v16bf b = b_frag_lds(smem, kt * 4 + nt);
            acc[nt] = __builtin_amdgcn_wmma_f32_16x16x32_bf16(
                false, a, false, b, (short)0, acc[nt], false, false);
        }
    }
    int l = threadIdx.x & 31;
    int n = l & 15;
    int mo = (l >> 4) * 8;
#pragma unroll
    for (int nt = 0; nt < 4; ++nt) {
        float bv = bias[nt * 16 + n];
#pragma unroll
        for (int r = 0; r < 8; ++r) {
            msg[(size_t)(row0 + mo + r) * OUT_DIM + nt * 16 + n] =
                acc[nt][r] + bv;
        }
    }
}

// ---------------------------------------------------------------------------
// Edge scatter: agg[dst] += msg[src]   (64 lanes per edge -> coalesced cols)
// Prefetch the next edge-group's msg row to hide gather latency behind the
// atomic stream (global_prefetch_b8).
// ---------------------------------------------------------------------------
__global__ void scatter_kernel(const int* __restrict__ ei,
                               const float* __restrict__ msg,
                               float* __restrict__ agg, int nEdges) {
    int t = blockIdx.x * blockDim.x + threadIdx.x;
    int e = t >> 6;
    int c = t & 63;
    if (e >= nEdges) return;
    int eNext = e + 256;
    if (eNext < nEdges) {
        int srcNext = ei[eNext];
        __builtin_prefetch(&msg[(size_t)srcNext * OUT_DIM + c], 0, 0);
    }
    int src = ei[e];            // edge_index[0][e]
    int dst = ei[nEdges + e];   // edge_index[1][e]
    atomicAdd(&agg[(size_t)dst * OUT_DIM + c],
              msg[(size_t)src * OUT_DIM + c]);
}

// ---------------------------------------------------------------------------
// GEMM 3: gi = [relu(agg) | x] @ w_ih.T + b_ih   [N,192]x[192,192] -> f32
// Weights staged in LDS (72KB); 6 K-chunks x 12 N-tiles per wave.
// ---------------------------------------------------------------------------
__global__ __launch_bounds__(256)
void gi_gemm_kernel(const float* __restrict__ agg,
                    const __bf16* __restrict__ xb,
                    const __bf16* __restrict__ wpk,
                    const float* __restrict__ bias,
                    float* __restrict__ gi, int rowblocks) {
    extern __shared__ __bf16 smem[];
    stage_weights(wpk, smem, GRU_IN * GRU_3H);
    int wave = (blockIdx.x * blockDim.x + threadIdx.x) >> 5;
    if (wave >= rowblocks) return;
    int row0 = wave * 16;
    v8f acc[12] = {};
#pragma unroll
    for (int kt = 0; kt < 6; ++kt) {
        v16bf a = (kt < 2)
                      ? a_frag_f32(agg, row0, OUT_DIM, kt * 32, true)
                      : a_frag_bf16(xb, row0, IN_DIM, (kt - 2) * 32);
#pragma unroll
        for (int nt = 0; nt < 12; ++nt) {
            v16bf b = b_frag_lds(smem, kt * 12 + nt);
            acc[nt] = __builtin_amdgcn_wmma_f32_16x16x32_bf16(
                false, a, false, b, (short)0, acc[nt], false, false);
        }
    }
    int l = threadIdx.x & 31;
    int n = l & 15;
    int mo = (l >> 4) * 8;
#pragma unroll
    for (int nt = 0; nt < 12; ++nt) {
        float bv = bias[nt * 16 + n];
#pragma unroll
        for (int r = 0; r < 8; ++r) {
            gi[(size_t)(row0 + mo + r) * GRU_3H + nt * 16 + n] =
                acc[nt][r] + bv;
        }
    }
}

// ---------------------------------------------------------------------------
// GEMM 4: gh = h @ w_hh.T + b_hh   [N,64]x[64,192] -> f32. LDS 24KB.
// ---------------------------------------------------------------------------
__global__ __launch_bounds__(256)
void gh_gemm_kernel(const float* __restrict__ h,
                    const __bf16* __restrict__ wpk,
                    const float* __restrict__ bias,
                    float* __restrict__ gh, int rowblocks) {
    extern __shared__ __bf16 smem[];
    stage_weights(wpk, smem, OUT_DIM * GRU_3H);
    int wave = (blockIdx.x * blockDim.x + threadIdx.x) >> 5;
    if (wave >= rowblocks) return;
    int row0 = wave * 16;
    v8f acc[12] = {};
#pragma unroll
    for (int kt = 0; kt < 2; ++kt) {
        v16bf a = a_frag_f32(h, row0, OUT_DIM, kt * 32, false);
#pragma unroll
        for (int nt = 0; nt < 12; ++nt) {
            v16bf b = b_frag_lds(smem, kt * 12 + nt);
            acc[nt] = __builtin_amdgcn_wmma_f32_16x16x32_bf16(
                false, a, false, b, (short)0, acc[nt], false, false);
        }
    }
    int l = threadIdx.x & 31;
    int n = l & 15;
    int mo = (l >> 4) * 8;
#pragma unroll
    for (int nt = 0; nt < 12; ++nt) {
        float bv = bias[nt * 16 + n];
#pragma unroll
        for (int r = 0; r < 8; ++r) {
            gh[(size_t)(row0 + mo + r) * GRU_3H + nt * 16 + n] =
                acc[nt][r] + bv;
        }
    }
}

// ---------------------------------------------------------------------------
// GRU gates: h = (1-z)*n + z*h, updates h (f32) in place + bf16 copy of h.
// ---------------------------------------------------------------------------
__device__ __forceinline__ float sigmoidf_(float x) {
    return 1.0f / (1.0f + __expf(-x));
}

__global__ void gru_gates_kernel(const float* __restrict__ gi,
                                 const float* __restrict__ gh,
                                 float* __restrict__ h,
                                 __bf16* __restrict__ outb, int total) {
    int t = blockIdx.x * blockDim.x + threadIdx.x;
    if (t >= total) return;
    int node = t >> 6;
    int j = t & 63;
    size_t base = (size_t)node * GRU_3H;
    float r  = sigmoidf_(gi[base + j] + gh[base + j]);
    float z  = sigmoidf_(gi[base + 64 + j] + gh[base + 64 + j]);
    float nn = tanhf(gi[base + 128 + j] + r * gh[base + 128 + j]);
    float hv = h[t];
    float hn = (1.0f - z) * nn + z * hv;
    h[t] = hn;
    outb[t] = (__bf16)hn;
}

// ---------------------------------------------------------------------------
// Launch
// ---------------------------------------------------------------------------
extern "C" void kernel_launch(void* const* d_in, const int* in_sizes, int n_in,
                              void* d_out, int out_size, void* d_ws,
                              size_t ws_size, hipStream_t stream) {
    const float* x      = (const float*)d_in[0];
    const int*   ei     = (const int*)d_in[1];
    const float* mlp_w  = (const float*)d_in[2];
    const float* mlp_b  = (const float*)d_in[3];
    const float* conv_w = (const float*)d_in[4];
    const float* conv_b = (const float*)d_in[5];
    const float* w_ih   = (const float*)d_in[6];
    const float* w_hh   = (const float*)d_in[7];
    const float* b_ih   = (const float*)d_in[8];
    const float* b_hh   = (const float*)d_in[9];
    float* h = (float*)d_out;   // [N, 64] final output, GRU state

    const int N = in_sizes[0] / IN_DIM;   // 50000 (divisible by 16)
    const int E = in_sizes[1] / 2;        // 800000
    const int rowblocks = N / 16;

    // workspace carve-out
    char* ws = (char*)d_ws;
    size_t off = 0;
    auto carve = [&](size_t bytes) {
        size_t o = off;
        off = (off + bytes + 255) & ~(size_t)255;
        return o;
    };
    __bf16* xb    = (__bf16*)(ws + carve((size_t)N * IN_DIM * 2));
    __bf16* outb  = (__bf16*)(ws + carve((size_t)N * OUT_DIM * 2));
    float*  msg   = (float*)(ws + carve((size_t)N * OUT_DIM * 4));
    float*  agg   = (float*)(ws + carve((size_t)N * OUT_DIM * 4));
    float*  gi    = (float*)(ws + carve((size_t)N * GRU_3H * 4));
    float*  gh    = (float*)(ws + carve((size_t)N * GRU_3H * 4));
    __bf16* wmlp  = (__bf16*)(ws + carve((size_t)IN_DIM * OUT_DIM * 2));
    __bf16* wconv = (__bf16*)(ws + carve((size_t)OUT_DIM * OUT_DIM * 2));
    __bf16* wihp  = (__bf16*)(ws + carve((size_t)GRU_IN * GRU_3H * 2));
    __bf16* whhp  = (__bf16*)(ws + carve((size_t)OUT_DIM * GRU_3H * 2));

    // --- weight repack (bf16 B-fragment tiles) ---
    pack_weights_kernel<<<(IN_DIM * OUT_DIM + 255) / 256, 256, 0, stream>>>(
        mlp_w, wmlp, IN_DIM, OUT_DIM, OUT_DIM, 0);
    pack_weights_kernel<<<(OUT_DIM * OUT_DIM + 255) / 256, 256, 0, stream>>>(
        conv_w, wconv, OUT_DIM, OUT_DIM, OUT_DIM, 0);
    pack_weights_kernel<<<(GRU_IN * GRU_3H + 255) / 256, 256, 0, stream>>>(
        w_ih, wihp, GRU_IN, GRU_3H, GRU_IN, 1);   // B[k][n] = w_ih[n][k]
    pack_weights_kernel<<<(OUT_DIM * GRU_3H + 255) / 256, 256, 0, stream>>>(
        w_hh, whhp, OUT_DIM, GRU_3H, OUT_DIM, 1); // B[k][n] = w_hh[n][k]

    // --- x -> bf16 ---
    cvt_f32_to_bf16_kernel<<<((size_t)N * IN_DIM + 255) / 256, 256, 0,
                             stream>>>(x, xb, N * IN_DIM);

    // --- MLP: h = x @ mlp_w + b ---
    int gemm_blocks = (rowblocks + 7) / 8;  // 8 waves / block
    mlp_gemm_kernel<<<gemm_blocks, 256, IN_DIM * OUT_DIM * 2, stream>>>(
        xb, wmlp, mlp_b, h, outb, rowblocks);

    for (int step = 0; step < 2; ++step) {
        conv_gemm_kernel<<<gemm_blocks, 256, OUT_DIM * OUT_DIM * 2, stream>>>(
            outb, wconv, conv_b, msg, rowblocks);
        zero_f32_kernel<<<((size_t)N * OUT_DIM + 255) / 256, 256, 0, stream>>>(
            agg, N * OUT_DIM);
        scatter_kernel<<<((size_t)E * OUT_DIM + 255) / 256, 256, 0, stream>>>(
            ei, msg, agg, E);
        gi_gemm_kernel<<<gemm_blocks, 256, GRU_IN * GRU_3H * 2, stream>>>(
            agg, xb, wihp, b_ih, gi, rowblocks);
        gh_gemm_kernel<<<gemm_blocks, 256, OUT_DIM * GRU_3H * 2, stream>>>(
            h, whhp, b_hh, gh, rowblocks);
        gru_gates_kernel<<<((size_t)N * OUT_DIM + 255) / 256, 256, 0, stream>>>(
            gi, gh, h, outb, N * OUT_DIM);
    }
}
